// LadderSideAdapterPruning_40896678592868
// MI455X (gfx1250) — matrix operand order repaired and checked by reading, not compiled, not run on hardware
//
#include <hip/hip_runtime.h>
#include <hip/hip_bf16.h>

#define KBLK   6
#define SEQ    257
#define BATCH  64
#define DMODEL 1024
#define HEADS  16
#define DRDIM  256
#define NROWS  (SEQ * BATCH)      // 16448
#define TINV   10.0f              // 1/T
#define LNEPS  1e-5f
#define POUT   768

#if defined(__AMDGCN__) && __has_builtin(__builtin_amdgcn_global_load_async_to_lds_b128) && __has_builtin(__builtin_amdgcn_s_wait_asynccnt)
#define USE_ASYNC 1
#else
#define USE_ASYNC 0
#endif

typedef __attribute__((ext_vector_type(16))) __bf16 v16bf;
typedef __attribute__((ext_vector_type(8)))  float  v8f;

#if USE_ASYNC
typedef int v4i_ __attribute__((vector_size(16)));
typedef __attribute__((address_space(1))) v4i_* gasp_t;   // global int4*
typedef __attribute__((address_space(3))) v4i_* lasp_t;   // LDS int4*
#endif

union FragU {
    unsigned short s[16];
    uint4          q[2];
    v16bf          v;
};

__device__ __forceinline__ unsigned short f2bf(float f) {
    unsigned u = __float_as_uint(f);
    u += 0x7FFFu + ((u >> 16) & 1u);          // round-to-nearest-even
    return (unsigned short)(u >> 16);
}
__device__ __forceinline__ unsigned pack2(float a, float b) {
    return (unsigned)f2bf(a) | ((unsigned)f2bf(b) << 16);
}

// ---------------------------------------------------------------------------
// x = w*res + (1-w)*x   (w = sigmoid(alpha[k]/T));  first block: x = res
// ---------------------------------------------------------------------------
__global__ void mix_kernel(const float* __restrict__ res,
                           const float* __restrict__ x_in,
                           float* __restrict__ x_out,
                           const float* __restrict__ alpha,
                           int k, int first, int n4) {
    int i = blockIdx.x * blockDim.x + threadIdx.x;
    if (i >= n4) return;
    float4 r = ((const float4*)res)[i];
    if (first) { ((float4*)x_out)[i] = r; return; }
    float w  = 1.0f / (1.0f + __expf(-alpha[k] * TINV));
    float iw = 1.0f - w;
    float4 x = ((const float4*)x_in)[i];
    float4 o;
    o.x = w * r.x + iw * x.x;
    o.y = w * r.y + iw * x.y;
    o.z = w * r.z + iw * x.z;
    o.w = w * r.w + iw * x.w;
    ((float4*)x_out)[i] = o;
}

// ---------------------------------------------------------------------------
// weight conversion: f32 -> bf16 (flat), and transposing variant for proj
// ---------------------------------------------------------------------------
__global__ void conv_kernel(const float* __restrict__ src,
                            unsigned short* __restrict__ dst, int n4) {
    int i = blockIdx.x * blockDim.x + threadIdx.x;
    if (i >= n4) return;
    float4 v = ((const float4*)src)[i];
    ((uint2*)dst)[i] = make_uint2(pack2(v.x, v.y), pack2(v.z, v.w));
}

__global__ void convT_kernel(const float* __restrict__ src,     // [1024][768]
                             unsigned short* __restrict__ dst)  // [768][1024]
{
    int i = blockIdx.x * blockDim.x + threadIdx.x;
    if (i >= POUT * DMODEL) return;
    int n = i >> 10, k = i & 1023;
    dst[i] = f2bf(src[(size_t)k * POUT + n]);
}

// ---------------------------------------------------------------------------
// LayerNorm over D=1024, one block (256 threads) per row; bf16 output
// ---------------------------------------------------------------------------
__global__ __launch_bounds__(256) void ln_kernel(const float* __restrict__ in,
                                                 const float* __restrict__ gw,
                                                 const float* __restrict__ gb,
                                                 unsigned short* __restrict__ out) {
    __shared__ float red[2][8];
    int row = blockIdx.x;
    const float4* p = (const float4*)(in + (size_t)row * DMODEL);
    float4 v = p[threadIdx.x];
    float s  = v.x + v.y + v.z + v.w;
    float ss = v.x * v.x + v.y * v.y + v.z * v.z + v.w * v.w;
#pragma unroll
    for (int m = 1; m < 32; m <<= 1) {
        s  += __shfl_xor(s, m, 32);
        ss += __shfl_xor(ss, m, 32);
    }
    int wave = threadIdx.x >> 5, lane = threadIdx.x & 31;
    if (lane == 0) { red[0][wave] = s; red[1][wave] = ss; }
    __syncthreads();
    float ts = 0.f, tss = 0.f;
#pragma unroll
    for (int i = 0; i < 8; ++i) { ts += red[0][i]; tss += red[1][i]; }
    float mu  = ts * (1.0f / DMODEL);
    float var = tss * (1.0f / DMODEL) - mu * mu;
    float rs  = rsqrtf(var + LNEPS);
    float4 w4 = ((const float4*)gw)[threadIdx.x];
    float4 b4 = ((const float4*)gb)[threadIdx.x];
    float ox = (v.x - mu) * rs * w4.x + b4.x;
    float oy = (v.y - mu) * rs * w4.y + b4.y;
    float oz = (v.z - mu) * rs * w4.z + b4.z;
    float ow = (v.w - mu) * rs * w4.w + b4.w;
    ((uint2*)(out + (size_t)row * DMODEL))[threadIdx.x] =
        make_uint2(pack2(ox, oy), pack2(oz, ow));
}

// ---------------------------------------------------------------------------
// WMMA GEMM: C[M,N] = A[M,K](bf16) * W[N,K](bf16)^T + bias.
// mode bits: 1=quickGELU, 2=+f32 residual, 4=transposed (b,s) output,
// 8=bf16 output.  Tiles 128x128x32; 8 waves; wave = 4(M) x 2(N) WMMA tiles.
// A/W tiles staged to LDS via async b128 load-to-LDS when available.
// ---------------------------------------------------------------------------
#define AST 40   // LDS row stride in ushorts (80B: 16B aligned, conflict-free)

__global__ __launch_bounds__(256) void gemm_kernel(
        const unsigned short* __restrict__ A, const unsigned short* __restrict__ W,
        const float* __restrict__ bias, const float* __restrict__ res,
        void* __restrict__ Cout, int M, int N, int Kd, int mode) {
    __shared__ unsigned short As[128 * AST];
    __shared__ unsigned short Ws[128 * AST];

    int m0 = blockIdx.y * 128;
    int n0 = blockIdx.x * 128;
    int tid = threadIdx.x;
    int wave = tid >> 5, lane = tid & 31;
    int wm = wave >> 2;          // 0..1
    int wn = wave & 3;           // 0..3
    int lcol = lane & 15;
    int g = lane >> 4;

    v8f acc[4][2];
#pragma unroll
    for (int mt = 0; mt < 4; ++mt)
#pragma unroll
        for (int nt = 0; nt < 2; ++nt)
            acc[mt][nt] = (v8f){0, 0, 0, 0, 0, 0, 0, 0};

    for (int k0 = 0; k0 < Kd; k0 += 32) {
        // ---- stage A / W tiles (128x32 bf16 = 512 x 16B each) ----
#if USE_ASYNC
#pragma unroll
        for (int p = 0; p < 2; ++p) {
            int id = tid + p * 256;
            int row = id >> 2;
            int c8 = (id & 3) * 8;
            int gm = m0 + row;
            if (gm < M)
                __builtin_amdgcn_global_load_async_to_lds_b128(
                    (gasp_t)(A + (size_t)gm * Kd + k0 + c8),
                    (lasp_t)(&As[row * AST + c8]), 0, 0);
            __builtin_amdgcn_global_load_async_to_lds_b128(
                (gasp_t)(W + (size_t)(n0 + row) * Kd + k0 + c8),
                (lasp_t)(&Ws[row * AST + c8]), 0, 0);
        }
#else
#pragma unroll
        for (int p = 0; p < 2; ++p) {
            int id = tid + p * 256;
            int row = id >> 2;
            int c8 = (id & 3) * 8;
            int gm = m0 + row;
            uint4 av = make_uint4(0, 0, 0, 0);
            if (gm < M) av = *(const uint4*)(A + (size_t)gm * Kd + k0 + c8);
            *(uint4*)&As[row * AST + c8] = av;
            *(uint4*)&Ws[row * AST + c8] =
                *(const uint4*)(W + (size_t)(n0 + row) * Kd + k0 + c8);
        }
#endif
        // prefetch next K tile (global_prefetch_b8)
        if (k0 + 32 < Kd) {
            int r = tid & 127;
            if (m0 + r < M) __builtin_prefetch(A + (size_t)(m0 + r) * Kd + k0 + 32, 0, 1);
            __builtin_prefetch(W + (size_t)(n0 + r) * Kd + k0 + 32, 0, 1);
        }
#if USE_ASYNC
        __builtin_amdgcn_s_wait_asynccnt(0);
#endif
        __syncthreads();

        // ---- fragments + WMMA ----
        FragU bfr[2];
#pragma unroll
        for (int nt = 0; nt < 2; ++nt) {
            const unsigned short* src = &Ws[(wn * 32 + nt * 16 + lcol) * AST + g * 8];
            bfr[nt].q[0] = *(const uint4*)(src);
            bfr[nt].q[1] = *(const uint4*)(src + 16);
        }
#pragma unroll
        for (int mt = 0; mt < 4; ++mt) {
            FragU afr;
            const unsigned short* src = &As[(wm * 64 + mt * 16 + lcol) * AST + g * 8];
            afr.q[0] = *(const uint4*)(src);
            afr.q[1] = *(const uint4*)(src + 16);
#pragma unroll
            for (int nt = 0; nt < 2; ++nt) {
                acc[mt][nt] = __builtin_amdgcn_wmma_f32_16x16x32_bf16(
                    false, afr.v, false, bfr[nt].v, (short)0, acc[mt][nt],
                    false, false);
            }
        }
        __syncthreads();
    }

    // ---- epilogue ----
#pragma unroll
    for (int mt = 0; mt < 4; ++mt)
#pragma unroll
        for (int nt = 0; nt < 2; ++nt) {
            int n = n0 + wn * 32 + nt * 16 + lcol;
            float bv = bias ? bias[n] : 0.0f;
#pragma unroll
            for (int v = 0; v < 8; ++v) {
                int m = m0 + wm * 64 + mt * 16 + v + 8 * g;
                if (m >= M) continue;
                float val = acc[mt][nt][v] + bv;
                if (mode & 1) val = val / (1.0f + __expf(-1.702f * val)); // quickGELU
                if (mode & 2) val += res[(size_t)m * N + n];
                size_t oidx;
                if (mode & 4) {
                    int b = m % BATCH, s = m / BATCH;
                    oidx = ((size_t)b * SEQ + s) * N + n;
                } else {
                    oidx = (size_t)m * N + n;
                }
                if (mode & 8) ((unsigned short*)Cout)[oidx] = f2bf(val);
                else          ((float*)Cout)[oidx] = val;
            }
        }
}

// ---------------------------------------------------------------------------
// Attention: one block (128 thr = 4 waves) per head (1024 heads), bf16 in/out.
// HD=16 -> WMMA K-dim 16..31 zero-padded: A hi-half zeros in registers,
// B hi-half reads a zeroed LDS row (branch-free, unconditional ds_load_b128).
// 1/sqrt(HD) folded onto f32 scores. Softmax f32; probs transposed through
// per-wave LDS to A-fragment layout; P*V in 9 WMMA K-steps (288 padded keys).
// ---------------------------------------------------------------------------
#define KST 24     // Ks row stride (ushorts, 48B)
#define VST 296    // Vt / Ps row stride (ushorts, 592B)

__global__ __launch_bounds__(128) void attn_kernel(
        const unsigned short* __restrict__ qkv, unsigned short* __restrict__ o) {
    __shared__ unsigned short Ks[272 * KST];
    __shared__ unsigned short Vt[16 * VST];
    __shared__ unsigned short Ps[4 * 16 * VST];
    __shared__ unsigned short Zs[32];

    int head = blockIdx.x;
    int b = head >> 4, h = head & 15;
    int tid = threadIdx.x;
    const unsigned short* base = qkv + (size_t)b * 768 + h * 16;

    // stage K rows (16B x 2 per row) and V transposed (Vt[e][s])
    for (int id = tid; id < 272 * 2; id += 128) {
        int s = id >> 1, c = (id & 1) * 8;
        uint4 val = make_uint4(0, 0, 0, 0);
        if (s < SEQ) val = *(const uint4*)(base + (size_t)s * BATCH * 768 + 256 + c);
        *(uint4*)&Ks[s * KST + c] = val;
    }
    for (int id = tid; id < 272 * 16; id += 128) {
        int s = id >> 4, e = id & 15;
        unsigned short vv = 0;
        if (s < SEQ) vv = base[(size_t)s * BATCH * 768 + 512 + e];
        Vt[e * VST + s] = vv;
    }
    for (int id = tid; id < 16 * 16; id += 128)      // zero Vt cols 272..287
        Vt[(id >> 4) * VST + 272 + (id & 15)] = 0;
    if (tid < 32) Zs[tid] = 0;
    __syncthreads();

    int wave = tid >> 5, lane = tid & 31;
    int lcol = lane & 15, g = lane >> 4;
    unsigned short* Pw = &Ps[wave * 16 * VST];

    // branch-free B-source: group-1 lanes read the zeroed row
    const unsigned short* kbase = g ? Zs : &Ks[lcol * KST];
    const int kstep = g ? 0 : 16 * KST;

    for (int qt = wave; qt < 17; qt += 4) {
        // Q A-fragment (16 bf16 = two 16B loads; lane half 1 = K-dim padding)
        FragU qf;
        int qrow = qt * 16 + lcol;
        if (qrow < SEQ && g == 0) {
            qf.q[0] = *(const uint4*)(base + (size_t)qrow * BATCH * 768);
            qf.q[1] = *(const uint4*)(base + (size_t)qrow * BATCH * 768 + 8);
        } else {
            qf.q[0] = make_uint4(0, 0, 0, 0);
            qf.q[1] = make_uint4(0, 0, 0, 0);
        }

        // scores: 17 tiles of 16x16
        v8f sc[17];
#pragma unroll
        for (int st = 0; st < 17; ++st) sc[st] = (v8f){0, 0, 0, 0, 0, 0, 0, 0};
#pragma unroll
        for (int st = 0; st < 17; ++st) {
            FragU kf;
            const unsigned short* src = kbase + st * kstep;
            kf.q[0] = *(const uint4*)(src);
            kf.q[1] = *(const uint4*)(src + 8);
            sc[st] = __builtin_amdgcn_wmma_f32_16x16x32_bf16(
                false, qf.v, false, kf.v, (short)0, sc[st], false, false);
        }
        // scale by 1/sqrt(HD) = 0.25
#pragma unroll
        for (int st = 0; st < 17; ++st)
#pragma unroll
            for (int v = 0; v < 8; ++v) sc[st][v] *= 0.25f;
        // mask padded keys in the last tile (s = 256 + lcol, valid only lcol==0)
        if (lcol > 0) {
#pragma unroll
            for (int v = 0; v < 8; ++v) sc[16][v] = -3.0e38f;
        }

        // row max over 272 keys (16-lane half-row reduction)
        v8f rmax = sc[0];
#pragma unroll
        for (int st = 1; st < 17; ++st)
#pragma unroll
            for (int v = 0; v < 8; ++v) rmax[v] = fmaxf(rmax[v], sc[st][v]);
#pragma unroll
        for (int m = 1; m < 16; m <<= 1)
#pragma unroll
            for (int v = 0; v < 8; ++v)
                rmax[v] = fmaxf(rmax[v], __shfl_xor(rmax[v], m, 32));

        // exp, row sum, transpose probs into per-wave LDS (A-fragment layout)
        v8f rsum = (v8f){0, 0, 0, 0, 0, 0, 0, 0};
#pragma unroll
        for (int st = 0; st < 17; ++st)
#pragma unroll
            for (int v = 0; v < 8; ++v) {
                float e = __expf(sc[st][v] - rmax[v]);   // masked -> 0
                rsum[v] += e;
                Pw[(v + 8 * g) * VST + st * 16 + lcol] = f2bf(e);
            }
#pragma unroll
        for (int j = 0; j < 8; ++j)                      // zero P cols 272..287
            Pw[lcol * VST + 272 + g * 8 + j] = 0;
#pragma unroll
        for (int m = 1; m < 16; m <<= 1)
#pragma unroll
            for (int v = 0; v < 8; ++v) rsum[v] += __shfl_xor(rsum[v], m, 32);

        // o = P * V  (K = 288 = 9 x 32)
        v8f oacc = (v8f){0, 0, 0, 0, 0, 0, 0, 0};
#pragma unroll
        for (int ks = 0; ks < 9; ++ks) {
            FragU pf;
            const unsigned short* ps = &Pw[lcol * VST + ks * 32 + g * 8];
            pf.q[0] = *(const uint4*)(ps);
            pf.q[1] = *(const uint4*)(ps + 16);
            FragU vf;
            const unsigned short* vs = &Vt[lcol * VST + ks * 32 + g * 8];
            vf.q[0] = *(const uint4*)(vs);
            vf.q[1] = *(const uint4*)(vs + 16);
            oacc = __builtin_amdgcn_wmma_f32_16x16x32_bf16(
                false, pf.v, false, vf.v, (short)0, oacc, false, false);
        }

        // normalize + store bf16 to (S, B, DR)
#pragma unroll
        for (int v = 0; v < 8; ++v) {
            int srow = qt * 16 + v + 8 * g;
            if (srow < SEQ)
                o[((size_t)srow * BATCH + b) * DRDIM + h * 16 + lcol] =
                    f2bf(oacc[v] / rsum[v]);
        }
    }
}

// ---------------------------------------------------------------------------
extern "C" void kernel_launch(void* const* d_in, const int* in_sizes, int n_in,
                              void* d_out, int out_size, void* d_ws, size_t ws_size,
                              hipStream_t stream) {
    const float* hook  = (const float*)d_in[0];
    const float* alpha = (const float*)d_in[1];
    const float* ipw   = (const float*)d_in[2];
    const float* ipb   = (const float*)d_in[3];
    const float* opw   = (const float*)d_in[4];
    const float* opb   = (const float*)d_in[5];
    const float* ln1w  = (const float*)d_in[6];
    const float* ln1b  = (const float*)d_in[7];
    const float* ln2w  = (const float*)d_in[8];
    const float* ln2b  = (const float*)d_in[9];
    const float* fcw   = (const float*)d_in[10];
    const float* fcb   = (const float*)d_in[11];
    const float* cw    = (const float*)d_in[12];
    const float* cb    = (const float*)d_in[13];
    const float* lnpw  = (const float*)d_in[14];
    const float* lnpb  = (const float*)d_in[15];
    const float* proj  = (const float*)d_in[16];

    const size_t rowsD = (size_t)NROWS * DMODEL;
    // workspace carve-up
    float*          x     = (float*)d_ws;                         // f32 residual
    unsigned short* h     = (unsigned short*)(x + rowsD);         // bf16
    unsigned short* t     = h + rowsD;                            // bf16
    unsigned short* attno = t + rowsD;                            // bf16
    unsigned short* wipw  = attno + (size_t)NROWS * DRDIM;
    unsigned short* wopw  = wipw + (size_t)KBLK * 768 * DMODEL;
    unsigned short* wfcw  = wopw + (size_t)KBLK * DMODEL * DRDIM;
    unsigned short* wcw   = wfcw + (size_t)KBLK * DMODEL * DMODEL;
    unsigned short* wproj = wcw  + (size_t)KBLK * DMODEL * DMODEL;
    float* out = (float*)d_out;

    // ---- weight conversion (amortized across 129 M-blocks per GEMM) ----
    {
        int n4;
        n4 = KBLK * 768 * DMODEL / 4;
        conv_kernel<<<(n4 + 255) / 256, 256, 0, stream>>>(ipw, wipw, n4);
        n4 = KBLK * DMODEL * DRDIM / 4;
        conv_kernel<<<(n4 + 255) / 256, 256, 0, stream>>>(opw, wopw, n4);
        n4 = KBLK * DMODEL * DMODEL / 4;
        conv_kernel<<<(n4 + 255) / 256, 256, 0, stream>>>(fcw, wfcw, n4);
        conv_kernel<<<(n4 + 255) / 256, 256, 0, stream>>>(cw, wcw, n4);
        convT_kernel<<<(POUT * DMODEL + 255) / 256, 256, 0, stream>>>(proj, wproj);
    }

    const int n4 = (int)(rowsD / 4);
    dim3 mixg((n4 + 255) / 256);
    const int mblk = (NROWS + 127) / 128;   // 129

    for (int i = 0; i < KBLK; ++i) {
        const float* res = hook + (size_t)i * rowsD;
        mix_kernel<<<mixg, 256, 0, stream>>>(res, x, x, alpha, i, (i == 0) ? 1 : 0, n4);

        ln_kernel<<<NROWS, 256, 0, stream>>>(x, ln1w + i * DMODEL, ln1b + i * DMODEL, h);

        // qkv(bf16) = h @ ipw^T + ipb   (M=16448, N=768, K=1024)
        gemm_kernel<<<dim3(768 / 128, mblk), 256, 0, stream>>>(
            h, wipw + (size_t)i * 768 * DMODEL, ipb + (size_t)i * 768,
            nullptr, t, NROWS, 768, DMODEL, 8);

        attn_kernel<<<BATCH * HEADS, 128, 0, stream>>>(t, attno);

        // x = x + attno @ opw^T + opb   (N=1024, K=256, f32 residual out)
        gemm_kernel<<<dim3(DMODEL / 128, mblk), 256, 0, stream>>>(
            attno, wopw + (size_t)i * DMODEL * DRDIM, opb + (size_t)i * DMODEL,
            x, x, NROWS, DMODEL, DRDIM, 2);

        ln_kernel<<<NROWS, 256, 0, stream>>>(x, ln2w + i * DMODEL, ln2b + i * DMODEL, h);

        // t(bf16) = quickGELU(h @ fcw^T + fcb)
        gemm_kernel<<<dim3(DMODEL / 128, mblk), 256, 0, stream>>>(
            h, wfcw + (size_t)i * DMODEL * DMODEL, fcb + (size_t)i * DMODEL,
            nullptr, t, NROWS, DMODEL, DMODEL, 1 | 8);

        // x = x + t @ cw^T + cb
        gemm_kernel<<<dim3(DMODEL / 128, mblk), 256, 0, stream>>>(
            t, wcw + (size_t)i * DMODEL * DMODEL, cb + (size_t)i * DMODEL,
            x, x, NROWS, DMODEL, DMODEL, 2);
    }

    // final LN + projection (wproj pre-transposed to [N=768][K=1024] bf16)
    ln_kernel<<<NROWS, 256, 0, stream>>>(x, lnpw, lnpb, h);
    gemm_kernel<<<dim3(POUT / 128, mblk), 256, 0, stream>>>(
        h, wproj, nullptr, nullptr, out, NROWS, POUT, DMODEL, 4);
}
